// CSNeuralODE_25907242730184
// MI455X (gfx1250) — matrix operand order repaired and checked by reading, not compiled, not run on hardware
//
#include <hip/hip_runtime.h>

// CDNA5 / gfx1250 Neural-ODE Euler integrator.
// B=4096, D=64, H=256, T=100.  One wave32 per block owns a 16-row batch tile
// and keeps y/u state in WMMA C-layout f32 registers for all 99 steps.  All
// four weight matrices live transposed (f16, [N][K]) in LDS; every GEMM is
// built from v_wmma_f32_16x16x32_f16 with f32 accumulation.  The two MLPs
// (f on y, g on u) are independent within a step and are computed interleaved
// so each one's WMMA/tanh latency is hidden by the other's instructions.

typedef __attribute__((ext_vector_type(16))) _Float16 v16h;
typedef __attribute__((ext_vector_type(8)))  _Float16 v8h;
typedef __attribute__((ext_vector_type(8)))  float    v8f;

#define BTOT   4096
#define DDIM   64
#define HDIM   256
#define TSTEPS 100
#define TWO_LOG2E 2.885390081777926815f  // 2*log2(e)

// A-matrix 16x32 f16 fragment (CDNA5 layout):
// lanes 0-15: v0-3 = K k0+0..7,  v4-7 = K k0+16..23   (M = lane)
// lanes16-31: v0-3 = K k0+8..15, v4-7 = K k0+24..31   (M = lane-16)
__device__ __forceinline__ v16h load_fragA(const _Float16* base, int ldk,
                                           int k0, int lane) {
  int m  = lane & 15;
  int hl = lane >> 4;
  const _Float16* p0 = base + m * ldk + k0 + hl * 8;
  v8h lo = *(const v8h*)p0;
  v8h hi = *(const v8h*)(p0 + 16);
  return __builtin_shufflevector(lo, hi, 0,1,2,3,4,5,6,7,8,9,10,11,12,13,14,15);
}

// B-matrix 32x16 f16 fragment: lane holds column N = nbase + (lane&15);
// lanes 0-15 hold K k0+0..7,16..23; lanes 16-31 hold K k0+8..15,24..31.
// Weights are stored transposed in LDS as [N][K] so this is contiguous.
__device__ __forceinline__ v16h load_fragB(const _Float16* wt, int ldk,
                                           int nbase, int k0, int lane) {
  int n  = nbase + (lane & 15);
  int hl = lane >> 4;
  const _Float16* p = wt + n * ldk + k0 + hl * 8;
  v8h lo = *(const v8h*)p;
  v8h hi = *(const v8h*)(p + 16);
  return __builtin_shufflevector(lo, hi, 0,1,2,3,4,5,6,7,8,9,10,11,12,13,14,15);
}

__global__ __launch_bounds__(32)
void CSNeuralODE_gfx1250_kernel(const float* __restrict__ y0,
                                const float* __restrict__ tg,
                                const float* __restrict__ Wf1,
                                const float* __restrict__ bf1,
                                const float* __restrict__ Wf2,
                                const float* __restrict__ bf2,
                                const float* __restrict__ Wg1,
                                const float* __restrict__ bg1,
                                const float* __restrict__ Wg2,
                                const float* __restrict__ bg2,
                                float* __restrict__ out) {
  extern __shared__ char smem[];
  _Float16* sWf1 = (_Float16*)smem;          // [256][64]  f16 (W^T)
  _Float16* sWf2 = sWf1 + HDIM * DDIM;       // [64][256]
  _Float16* sWg1 = sWf2 + DDIM * HDIM;       // [256][64]
  _Float16* sWg2 = sWg1 + HDIM * DDIM;       // [64][256]
  float* sb   = (float*)(sWg2 + DDIM * HDIM);
  float* sbf1 = sb;          // 256 (pre-scaled by 2log2e)
  float* sbf2 = sb + 256;    // 64
  float* sbg1 = sb + 320;    // 256 (pre-scaled by 2log2e)
  float* sbg2 = sb + 576;    // 64
  float* st   = sb + 640;    // 100 (padded to 128)
  _Float16* stAy = (_Float16*)(st + 128);    // 16*64  f16 staging (y)
  _Float16* stAu = stAy + 16 * DDIM;         // 16*64  f16 staging (u)
  _Float16* hAy  = stAu + 16 * DDIM;         // 16*256 f16 staging (tanh, f)
  _Float16* hAu  = hAy  + 16 * HDIM;         // 16*256 f16 staging (tanh, g)

  const int tid  = threadIdx.x;   // blockDim == 32: one wave
  const int lane = tid & 31;

  // One-time: load weights from HBM, transpose to [N][K], convert to f16.
  for (int e = tid; e < DDIM * HDIM; e += 32) {
    int k = e >> 8, n = e & 255;               // Wf1/Wg1 are [64(K)][256(N)]
    sWf1[n * DDIM + k] = (_Float16)Wf1[e];
    sWg1[n * DDIM + k] = (_Float16)Wg1[e];
  }
  for (int e = tid; e < HDIM * DDIM; e += 32) {
    int k = e >> 6, n = e & 63;                // Wf2/Wg2 are [256(K)][64(N)]
    sWf2[n * HDIM + k] = (_Float16)Wf2[e];
    sWg2[n * HDIM + k] = (_Float16)Wg2[e];
  }
  for (int e = tid; e < HDIM; e += 32) {
    sbf1[e] = bf1[e] * TWO_LOG2E;
    sbg1[e] = bg1[e] * TWO_LOG2E;
  }
  for (int e = tid; e < DDIM; e += 32) { sbf2[e] = bf2[e]; sbg2[e] = bg2[e]; }
  for (int e = tid; e < TSTEPS; e += 32) st[e] = tg[e];
  __syncthreads();

  const int b0  = blockIdx.x * 16;
  const int col = lane & 15;
  const int hl  = lane >> 4;

  // Load y0 tile into C-layout registers; u0 = y0; emit t=0 output.
  v8f y[4], u[4];
#pragma unroll
  for (int n = 0; n < 4; ++n)
#pragma unroll
    for (int r = 0; r < 8; ++r) {
      size_t row = (size_t)(b0 + r + 8 * hl);
      float v = y0[row * DDIM + n * 16 + col];
      y[n][r] = v;
      u[n][r] = v;
      out[row * (TSTEPS * DDIM) + n * 16 + col] = v;
    }

  for (int s = 0; s < TSTEPS - 1; ++s) {
    float dt = st[s + 1] - st[s];

    // Stage old y and old u as f16 A-sources (both MLPs read OLD state).
#pragma unroll
    for (int n = 0; n < 4; ++n)
#pragma unroll
      for (int r = 0; r < 8; ++r) {
        int idx = (r + 8 * hl) * DDIM + n * 16 + col;
        stAy[idx] = (_Float16)y[n][r];
        stAu[idx] = (_Float16)u[n][r];
      }
    v16h ay0 = load_fragA(stAy, DDIM, 0,  lane);
    v16h ay1 = load_fragA(stAy, DDIM, 32, lane);
    v16h au0 = load_fragA(stAu, DDIM, 0,  lane);
    v16h au1 = load_fragA(stAu, DDIM, 32, lane);

    // GEMM1 x2 [16,64]x[64,256] + tanh -> hAy/hAu.  The two accumulator
    // chains and two 8-wide tanh batches are independent, filling WMMA->VALU
    // hazard slots and exp/rcp latency with real work.
#pragma unroll
    for (int n = 0; n < 16; ++n) {
      float bcY = sbf1[n * 16 + col];
      float bcU = sbg1[n * 16 + col];
      v16h bY0 = load_fragB(sWf1, DDIM, n * 16, 0,  lane);
      v16h bY1 = load_fragB(sWf1, DDIM, n * 16, 32, lane);
      v16h bU0 = load_fragB(sWg1, DDIM, n * 16, 0,  lane);
      v16h bU1 = load_fragB(sWg1, DDIM, n * 16, 32, lane);
      v8f aY = {}, aU = {};
      aY = __builtin_amdgcn_wmma_f32_16x16x32_f16(false, ay0, false, bY0,
                                                  (short)0, aY, false, false);
      aU = __builtin_amdgcn_wmma_f32_16x16x32_f16(false, au0, false, bU0,
                                                  (short)0, aU, false, false);
      aY = __builtin_amdgcn_wmma_f32_16x16x32_f16(false, ay1, false, bY1,
                                                  (short)0, aY, false, false);
      aU = __builtin_amdgcn_wmma_f32_16x16x32_f16(false, au1, false, bU1,
                                                  (short)0, aU, false, false);
      // tanh(x+b) = 1 - 2/(exp2(x*2log2e + b*2log2e) + 1), phased for ILP.
      float eY[8], eU[8];
#pragma unroll
      for (int r = 0; r < 8; ++r) eY[r] = aY[r] * TWO_LOG2E + bcY;
#pragma unroll
      for (int r = 0; r < 8; ++r) eU[r] = aU[r] * TWO_LOG2E + bcU;
#pragma unroll
      for (int r = 0; r < 8; ++r) eY[r] = __builtin_amdgcn_exp2f(eY[r]);
#pragma unroll
      for (int r = 0; r < 8; ++r) eU[r] = __builtin_amdgcn_exp2f(eU[r]);
#pragma unroll
      for (int r = 0; r < 8; ++r) eY[r] = __builtin_amdgcn_rcpf(eY[r] + 1.0f);
#pragma unroll
      for (int r = 0; r < 8; ++r) eU[r] = __builtin_amdgcn_rcpf(eU[r] + 1.0f);
#pragma unroll
      for (int r = 0; r < 8; ++r) {
        int idx = (r + 8 * hl) * HDIM + n * 16 + col;
        hAy[idx] = (_Float16)(1.0f - 2.0f * eY[r]);
        hAu[idx] = (_Float16)(1.0f - 2.0f * eU[r]);
      }
    }

    // GEMM2 x2 [16,256]x[256,64] -> fdot/udot (f32 C-layout).
    v8f dY[4], dU[4];
#pragma unroll
    for (int n = 0; n < 4; ++n) { dY[n] = (v8f){}; dU[n] = (v8f){}; }
#pragma unroll
    for (int k = 0; k < 8; ++k) {
      v16h aY = load_fragA(hAy, HDIM, k * 32, lane);
      v16h aU = load_fragA(hAu, HDIM, k * 32, lane);
#pragma unroll
      for (int n = 0; n < 4; ++n) {
        v16h bY = load_fragB(sWf2, HDIM, n * 16, k * 32, lane);
        dY[n] = __builtin_amdgcn_wmma_f32_16x16x32_f16(false, aY, false, bY,
                                                       (short)0, dY[n], false, false);
        v16h bU = load_fragB(sWg2, HDIM, n * 16, k * 32, lane);
        dU[n] = __builtin_amdgcn_wmma_f32_16x16x32_f16(false, aU, false, bU,
                                                       (short)0, dU[n], false, false);
      }
    }

    // Euler update + stream y_{s+1} to HBM.
#pragma unroll
    for (int n = 0; n < 4; ++n) {
      v8f ud = dU[n] + sbg2[n * 16 + col];
      v8f fd = dY[n] + sbf2[n * 16 + col];
      u[n] = u[n] + ud * dt;
      y[n] = y[n] + (fd + ud) * dt;
#pragma unroll
      for (int r = 0; r < 8; ++r) {
        size_t row = (size_t)(b0 + r + 8 * hl);
        out[row * (TSTEPS * DDIM) + (size_t)(s + 1) * DDIM + n * 16 + col] = y[n][r];
      }
    }
  }
}

extern "C" void kernel_launch(void* const* d_in, const int* in_sizes, int n_in,
                              void* d_out, int out_size, void* d_ws, size_t ws_size,
                              hipStream_t stream) {
  const float* y0  = (const float*)d_in[0];
  const float* tg  = (const float*)d_in[1];
  const float* Wf1 = (const float*)d_in[2];
  const float* bf1 = (const float*)d_in[3];
  const float* Wf2 = (const float*)d_in[4];
  const float* bf2 = (const float*)d_in[5];
  const float* Wg1 = (const float*)d_in[6];
  const float* bg1 = (const float*)d_in[7];
  const float* Wg2 = (const float*)d_in[8];
  const float* bg2 = (const float*)d_in[9];
  float* out = (float*)d_out;

  // LDS: 4 weight mats f16 (128 KB) + biases/t-grid (3 KB) + staging (20 KB).
  size_t shmem = (size_t)4 * HDIM * DDIM * 2                 // weights f16
               + (640 + 128) * sizeof(float)                 // biases + t grid
               + (size_t)2 * (16 * DDIM + 16 * HDIM) * 2;    // y/u staging

  dim3 grid(BTOT / 16);   // 256 blocks, one 16-row tile each
  dim3 block(32);         // one wave32
  CSNeuralODE_gfx1250_kernel<<<grid, block, shmem, stream>>>(
      y0, tg, Wf1, bf1, Wf2, bf2, Wg1, bg1, Wg2, bg2, out);
}
